// RelativeTransformer_58145267254160
// MI455X (gfx1250) — compile-verified, matrix operand order checked by reading, not run on hardware
//
#include <hip/hip_runtime.h>
#include <hip/hip_bf16.h>

// ---------------------------------------------------------------------------
// Problem constants (match reference)
// ---------------------------------------------------------------------------
#define S_LEN   1024
#define DMODEL  1024
#define NHEAD   16
#define HDIM    64
#define NLAYER  4
#define DFF_    4096
#define BATCH   2
#define NROWS   (BATCH * S_LEN)        // 2048
#define NUM_REL_ (2 * S_LEN - 1)       // 2047
#define LN_EPS  1e-5f

// ---------------------------------------------------------------------------
// CDNA5 WMMA / TDM types
// ---------------------------------------------------------------------------
typedef __attribute__((ext_vector_type(16))) __bf16 v16bf;
typedef __attribute__((ext_vector_type(8)))  __bf16 v8bf;
typedef __attribute__((ext_vector_type(8)))  float  v8f;
typedef __attribute__((ext_vector_type(4)))  unsigned v4u;
typedef __attribute__((ext_vector_type(8)))  int      v8i;
typedef __attribute__((ext_vector_type(4)))  int      v4i;

union Frag16 { v16bf v; v8bf h[2]; };
union Pk4 { __bf16 h[4]; uint2 u; };

// ---------------------------------------------------------------------------
// Generic bf16-WMMA GEMM:  C = act(A[M,K] @ W[N,K]^T + bias[N])
//   MODE 0: plain fp32 store [M,N]
//   MODE 1: exact GELU fp32 store [M,N]
//   MODE 2: permuted bf16 store to [B,H,S,HD]; val = val*alpha + ebar[s,d]
// Block: 256 threads (8 waves), tile 128x128, k-step 32.
// ---------------------------------------------------------------------------
#define TM 128
#define TN 128
#define TK 32
#define LDP 40          // padded LDS row stride (bf16 elems), 80B = 5*16B

template<int MODE>
__global__ __launch_bounds__(256) void gemm_bf16_wmma(
    const float* __restrict__ A,     // [M,K] fp32
    const float* __restrict__ W,     // [N,K] fp32
    const float* __restrict__ bias,  // [N]
    float* __restrict__ C,           // modes 0/1
    __bf16* __restrict__ Cbf,        // mode 2
    int M, int N, int K,
    float alpha, const float* __restrict__ ebar)
{
    __shared__ __align__(16) __bf16 sA[TM][LDP];
    __shared__ __align__(16) __bf16 sB[TN][LDP];

    const int tid  = threadIdx.x;
    const int wave = tid >> 5;
    const int lane = tid & 31;
    const int wm   = (wave >> 2) * 64;   // 0 / 64
    const int wn   = (wave & 3) * 32;    // 0..96
    const int bm   = blockIdx.y * TM;
    const int bn   = blockIdx.x * TN;

    v8f acc[4][2];
    for (int i = 0; i < 4; ++i)
        for (int j = 0; j < 2; ++j)
            for (int r = 0; r < 8; ++r) acc[i][j][r] = 0.0f;

    const int rL = tid >> 3;          // 0..31
    const int cL = (tid & 7) * 4;     // 0..28
    const int lr = lane & 15;
    const int hb = (lane >> 4) * 8;   // K-stripe base per lane-half

    for (int kt = 0; kt < K; kt += TK) {
        // ---- stage A tile (fp32 -> bf16, packed 8B stores) ----
        for (int it = 0; it < 4; ++it) {
            const int row = rL + it * 32;
            const float4 v = *(const float4*)&A[(size_t)(bm + row) * K + kt + cL];
            Pk4 p;
            p.h[0] = (__bf16)v.x; p.h[1] = (__bf16)v.y;
            p.h[2] = (__bf16)v.z; p.h[3] = (__bf16)v.w;
            *(uint2*)&sA[row][cL] = p.u;
        }
        // ---- stage W tile ----
        for (int it = 0; it < 4; ++it) {
            const int row = rL + it * 32;
            const float4 v = *(const float4*)&W[(size_t)(bn + row) * K + kt + cL];
            Pk4 p;
            p.h[0] = (__bf16)v.x; p.h[1] = (__bf16)v.y;
            p.h[2] = (__bf16)v.z; p.h[3] = (__bf16)v.w;
            *(uint2*)&sB[row][cL] = p.u;
        }
        __syncthreads();

        Frag16 af[4], bf[2];
        for (int i = 0; i < 4; ++i) {
            const __bf16* p = &sA[wm + i * 16 + lr][0];
            af[i].h[0] = *(const v8bf*)(p + hb);
            af[i].h[1] = *(const v8bf*)(p + 16 + hb);
        }
        for (int j = 0; j < 2; ++j) {
            const __bf16* p = &sB[wn + j * 16 + lr][0];
            bf[j].h[0] = *(const v8bf*)(p + hb);
            bf[j].h[1] = *(const v8bf*)(p + 16 + hb);
        }
        for (int i = 0; i < 4; ++i)
            for (int j = 0; j < 2; ++j)
                acc[i][j] = __builtin_amdgcn_wmma_f32_16x16x32_bf16(
                    false, af[i].v, false, bf[j].v, (short)0, acc[i][j], false, false);
        __syncthreads();
    }

    // ---- epilogue ----
    const int lh = lane >> 4;
    for (int i = 0; i < 4; ++i) {
        for (int j = 0; j < 2; ++j) {
            const int gm0 = bm + wm + i * 16 + lh * 8;
            const int gn  = bn + wn + j * 16 + lr;
            const float bv = bias[gn];
            for (int r = 0; r < 8; ++r) {
                const int gm = gm0 + r;
                float val = acc[i][j][r] + bv;
                if (MODE == 1) {
                    val = 0.5f * val * (1.0f + erff(val * 0.70710678118654752f));
                }
                if (MODE == 2) {
                    const int b = gm >> 10;        // / S_LEN
                    const int s = gm & 1023;
                    const int h = gn >> 6;         // / HDIM
                    const int d = gn & 63;
                    float e = ebar ? ebar[s * HDIM + d] : 0.0f;
                    val = val * alpha + e;
                    Cbf[(((size_t)b * NHEAD + h) * S_LEN + s) * HDIM + d] = (__bf16)val;
                } else {
                    C[(size_t)gm * N + gn] = val;
                }
            }
        }
    }
}

// ---------------------------------------------------------------------------
// Flash attention over Q·Keff^T (scale & rel-bias folded into Keff).
// Q/Keff/V are bf16 [B,H,S,HD]. Block: 128 threads (4 waves), each wave
// owns 16 query rows. Keff tiles staged to LDS by the Tensor Data Mover
// (tensor_load_to_lds + s_wait_tensorcnt); V staged transposed manually.
// ---------------------------------------------------------------------------
__global__ __launch_bounds__(128) void flash_attn_wmma(
    const __bf16* __restrict__ Q,   // [B,H,S,HD]
    const __bf16* __restrict__ Kf,  // [B,H,S,HD]  (Keff)
    const __bf16* __restrict__ V,   // [B,H,S,HD]
    float* __restrict__ O)          // [B,S,DM] fp32
{
    __shared__ __align__(16) __bf16 sK [32][72];   // TDM dest: 64 elem rows + 8 pad
    __shared__ __align__(16) __bf16 sVt[64][40];   // [d][j_local]
    __shared__ __align__(16) __bf16 sP [4][16][40];// per-wave P transpose scratch

    const int tid  = threadIdx.x;
    const int wave = tid >> 5;
    const int lane = tid & 31;
    const int lr   = lane & 15;
    const int lh   = lane >> 4;
    const int bh   = blockIdx.y;
    const int b    = bh >> 4;
    const int h    = bh & 15;
    const size_t base = (size_t)bh * S_LEN * HDIM;
    const int rowBase = blockIdx.x * 64 + wave * 16;

    // --- load Q fragments (A layout, 2 k-chunks of 32 over HD=64) ---
    Frag16 qf[2];
    {
        const __bf16* qrow = &Q[base + (size_t)(rowBase + lr) * HDIM];
        for (int kc = 0; kc < 2; ++kc) {
            const int b0 = kc * 32 + lh * 8;
            qf[kc].h[0] = *(const v8bf*)(qrow + b0);
            qf[kc].h[1] = *(const v8bf*)(qrow + b0 + 16);
        }
    }

    float rm[8], rl[8];
    v8f o[4];
    for (int r = 0; r < 8; ++r) { rm[r] = -1e30f; rl[r] = 0.0f; }
    for (int t = 0; t < 4; ++t)
        for (int r = 0; r < 8; ++r) o[t][r] = 0.0f;

    for (int jt = 0; jt < S_LEN; jt += 32) {
        // ---- TDM: stage Keff tile [32 x 64] bf16 with +16B row padding ----
        if (wave == 0) {
            const unsigned ldsK = (unsigned)(uintptr_t)&sK[0][0];
            const unsigned long long ga =
                (unsigned long long)(uintptr_t)(Kf + base + (size_t)jt * HDIM);
            v4u g0;
            g0[0] = 1u;                                  // count=1 valid descriptor
            g0[1] = ldsK;                                // lds_addr
            g0[2] = (unsigned)ga;                        // global_addr[31:0]
            g0[3] = (unsigned)(ga >> 32) | 0x80000000u;  // addr[56:32] | type=2
            v8i g1;
            g1[0] = (int)((1u << 16)    // data_size = 2 bytes
                        | (1u << 20)    // pad_enable
                        | (4u << 22)    // pad_interval: 32 dwords (=128B row)
                        | (3u << 25));  // pad_amount: 4 dwords (=16B)
            g1[1] = (int)(64u << 16);                 // tensor_dim0 = 64 (lo16)
            g1[2] = 0;                                // dim0 hi / dim1 lo
            g1[3] = (int)(16u | (64u << 16));         // dim1 hi (dim1=1<<20), tile_dim0=64
            g1[4] = 32;                               // tile_dim1 = 32, tile_dim2 = 0
            g1[5] = 64;                               // tensor_dim0_stride = 64 (lo32)
            g1[6] = (int)(64u << 16);                 // stride0 hi, tensor_dim1_stride lo16 = 64
            g1[7] = 0;
            v4i gz = {0, 0, 0, 0};
#if __clang_major__ >= 23
            v8i gz8 = {0, 0, 0, 0, 0, 0, 0, 0};
            __builtin_amdgcn_tensor_load_to_lds(g0, g1, gz, gz, gz8, 0);
#else
            __builtin_amdgcn_tensor_load_to_lds(g0, g1, gz, gz, 0);
#endif
        }
        // ---- stage V tile transposed: sVt[d][j] (bf16 copy, no convert) ----
        {
            const int j  = tid >> 2;          // 0..31
            const int c8 = (tid & 3) * 16;    // 0,16,32,48
            const __bf16* vrow = &V[base + (size_t)(jt + j) * HDIM + c8];
            v8bf a0 = *(const v8bf*)(vrow);
            v8bf a1 = *(const v8bf*)(vrow + 8);
            for (int i = 0; i < 8; ++i) sVt[c8 + i][j] = a0[i];
            for (int i = 0; i < 8; ++i) sVt[c8 + 8 + i][j] = a1[i];
        }
        if (wave == 0) __builtin_amdgcn_s_wait_tensorcnt(0);
        __syncthreads();

        // ---- scores: S[16x32] = Q[16x64] @ Keff_chunk^T[64x32] ----
        v8f s0, s1;
        for (int r = 0; r < 8; ++r) { s0[r] = 0.0f; s1[r] = 0.0f; }
        for (int kc = 0; kc < 2; ++kc) {
            const int off = kc * 32 + lh * 8;
            Frag16 bk0, bk1;
            { const __bf16* p = &sK[lr][0];
              bk0.h[0] = *(const v8bf*)(p + off);
              bk0.h[1] = *(const v8bf*)(p + off + 16); }
            { const __bf16* p = &sK[16 + lr][0];
              bk1.h[0] = *(const v8bf*)(p + off);
              bk1.h[1] = *(const v8bf*)(p + off + 16); }
            s0 = __builtin_amdgcn_wmma_f32_16x16x32_bf16(false, qf[kc].v, false, bk0.v, (short)0, s0, false, false);
            s1 = __builtin_amdgcn_wmma_f32_16x16x32_bf16(false, qf[kc].v, false, bk1.v, (short)0, s1, false, false);
        }

        // ---- online softmax (row = vgpr index + lane-half*8) ----
        float cm[8];
        for (int r = 0; r < 8; ++r) cm[r] = fmaxf(s0[r], s1[r]);
        for (int off = 1; off < 16; off <<= 1)
            for (int r = 0; r < 8; ++r)
                cm[r] = fmaxf(cm[r], __shfl_xor(cm[r], off, 32));

        float corr[8], cs[8];
        for (int r = 0; r < 8; ++r) {
            const float mn = fmaxf(rm[r], cm[r]);
            corr[r] = __expf(rm[r] - mn);
            s0[r] = __expf(s0[r] - mn);
            s1[r] = __expf(s1[r] - mn);
            rm[r] = mn;
            cs[r] = s0[r] + s1[r];
        }
        for (int off = 1; off < 16; off <<= 1)
            for (int r = 0; r < 8; ++r)
                cs[r] += __shfl_xor(cs[r], off, 32);
        for (int r = 0; r < 8; ++r) rl[r] = rl[r] * corr[r] + cs[r];
        for (int t = 0; t < 4; ++t)
            for (int r = 0; r < 8; ++r) o[t][r] *= corr[r];

        // ---- transpose P (D-layout -> A-layout) through LDS ----
        for (int r = 0; r < 8; ++r) {
            const int ri = r + lh * 8;
            sP[wave][ri][lr]      = (__bf16)s0[r];
            sP[wave][ri][16 + lr] = (__bf16)s1[r];
        }
        asm volatile("s_wait_dscnt 0" ::: "memory");
        Frag16 pf;
        { const __bf16* p = &sP[wave][lr][0];
          pf.h[0] = *(const v8bf*)(p + lh * 8);
          pf.h[1] = *(const v8bf*)(p + 16 + lh * 8); }

        // ---- O += P[16x32] @ V[32x64] ----
        for (int t = 0; t < 4; ++t) {
            Frag16 vf;
            const __bf16* p = &sVt[t * 16 + lr][0];
            vf.h[0] = *(const v8bf*)(p + lh * 8);
            vf.h[1] = *(const v8bf*)(p + 16 + lh * 8);
            o[t] = __builtin_amdgcn_wmma_f32_16x16x32_bf16(false, pf.v, false, vf.v, (short)0, o[t], false, false);
        }
        __syncthreads();
    }

    // ---- epilogue: divide by row-sum, store into [B,S,DM] ----
    for (int r = 0; r < 8; ++r) {
        const float inv = 1.0f / rl[r];
        const int row = rowBase + r + lh * 8;
        for (int t = 0; t < 4; ++t) {
            const int d = t * 16 + lr;
            O[((size_t)b * S_LEN + row) * DMODEL + h * HDIM + d] = o[t][r] * inv;
        }
    }
}

// ---------------------------------------------------------------------------
// Fused residual-add + LayerNorm (in place on xc). One block per row.
// ---------------------------------------------------------------------------
__global__ __launch_bounds__(256) void add_layernorm(
    float* __restrict__ xc, const float* __restrict__ y,
    const float* __restrict__ g, const float* __restrict__ be)
{
    __shared__ float red[2][8];
    const int row = blockIdx.x;
    const int tid = threadIdx.x;
    const int wave = tid >> 5, lane = tid & 31;
    float* xr = &xc[(size_t)row * DMODEL];
    const float* yr = &y[(size_t)row * DMODEL];

    float v[4]; float s = 0.f, s2 = 0.f;
    for (int i = 0; i < 4; ++i) {
        const int c = tid + i * 256;
        v[i] = xr[c] + yr[c];
        s += v[i]; s2 += v[i] * v[i];
    }
    for (int off = 16; off; off >>= 1) {
        s  += __shfl_xor(s,  off, 32);
        s2 += __shfl_xor(s2, off, 32);
    }
    if (lane == 0) { red[0][wave] = s; red[1][wave] = s2; }
    __syncthreads();
    if (wave == 0) {
        s  = (lane < 8) ? red[0][lane] : 0.f;
        s2 = (lane < 8) ? red[1][lane] : 0.f;
        for (int off = 4; off; off >>= 1) {
            s  += __shfl_xor(s,  off, 32);
            s2 += __shfl_xor(s2, off, 32);
        }
        if (lane == 0) { red[0][0] = s; red[1][0] = s2; }
    }
    __syncthreads();
    const float mean = red[0][0] * (1.0f / DMODEL);
    const float var  = red[1][0] * (1.0f / DMODEL) - mean * mean;
    const float rstd = rsqrtf(var + LN_EPS);
    for (int i = 0; i < 4; ++i) {
        const int c = tid + i * 256;
        xr[c] = (v[i] - mean) * rstd * g[c] + be[c];
    }
}

// ---------------------------------------------------------------------------
// rel prefix-sum over rows (per column d) + windowed mean -> Ebar[S,HD]
// ---------------------------------------------------------------------------
__global__ void rel_prefix(const float* __restrict__ rel, float* __restrict__ P)
{
    const int d = threadIdx.x;  // 64 threads
    float run = 0.f;
    P[d] = 0.f;
    for (int r = 0; r < NUM_REL_; ++r) {
        run += rel[(size_t)r * HDIM + d];
        P[(size_t)(r + 1) * HDIM + d] = run;
    }
}

__global__ void ebar_from_prefix(const float* __restrict__ P, float* __restrict__ E)
{
    const int idx = blockIdx.x * 256 + threadIdx.x;   // S*HD = 65536
    const int j = idx >> 6, d = idx & 63;
    // Ebar[j,d] = mean_k rel[k - j + S - 1, d] = (P[2S-1-j] - P[S-1-j]) / S
    E[idx] = (P[(size_t)(2 * S_LEN - 1 - j) * HDIM + d] -
              P[(size_t)(S_LEN - 1 - j) * HDIM + d]) * (1.0f / (float)S_LEN);
}

__global__ void copy_f32(const float* __restrict__ src, float* __restrict__ dst, int n)
{
    const int i = blockIdx.x * 256 + threadIdx.x;
    if (i < n) dst[i] = src[i];
}

// ---------------------------------------------------------------------------
// Host orchestration
// ---------------------------------------------------------------------------
extern "C" void kernel_launch(void* const* d_in, const int* in_sizes, int n_in,
                              void* d_out, int out_size, void* d_ws, size_t ws_size,
                              hipStream_t stream) {
    (void)in_sizes; (void)n_in; (void)out_size; (void)ws_size;

    const float* x    = (const float*)d_in[0];
    const float* Wq   = (const float*)d_in[1];
    const float* bq   = (const float*)d_in[2];
    const float* Wk   = (const float*)d_in[3];
    const float* bk   = (const float*)d_in[4];
    const float* Wv   = (const float*)d_in[5];
    const float* bv   = (const float*)d_in[6];
    const float* Wo   = (const float*)d_in[7];
    const float* bo   = (const float*)d_in[8];
    const float* rel  = (const float*)d_in[9];
    const float* W1   = (const float*)d_in[10];
    const float* b1   = (const float*)d_in[11];
    const float* W2   = (const float*)d_in[12];
    const float* b2   = (const float*)d_in[13];
    const float* g1   = (const float*)d_in[14];
    const float* be1  = (const float*)d_in[15];
    const float* g2   = (const float*)d_in[16];
    const float* be2  = (const float*)d_in[17];
    float* out = (float*)d_out;

    char* ws = (char*)d_ws;
    size_t off = 0;
    auto alloc = [&](size_t bytes) -> void* {
        void* p = (void*)(ws + off);
        off += (bytes + 255) & ~(size_t)255;
        return p;
    };
    float*  xc    = (float*)alloc((size_t)NROWS * DMODEL * 4);   //  8 MB
    __bf16* qbuf  = (__bf16*)alloc((size_t)NROWS * DMODEL * 2);  //  4 MB [B,H,S,HD]
    __bf16* kbuf  = (__bf16*)alloc((size_t)NROWS * DMODEL * 2);  //  4 MB
    __bf16* vbuf  = (__bf16*)alloc((size_t)NROWS * DMODEL * 2);  //  4 MB
    float*  attno = (float*)alloc((size_t)NROWS * DMODEL * 4);   //  8 MB [B,S,DM]
    float*  tmp   = (float*)alloc((size_t)NROWS * DMODEL * 4);   //  8 MB
    float*  hbuf  = (float*)alloc((size_t)NROWS * DFF_   * 4);   // 32 MB
    float*  Pbuf  = (float*)alloc((size_t)(NUM_REL_ + 1) * HDIM * 4);
    float*  ebar  = (float*)alloc((size_t)S_LEN * HDIM * 4);

    const dim3 blk256(256), blk128(128), blk64(64);
    const dim3 gemm_g_dm(DMODEL / TN, NROWS / TM);      // (8,16)
    const dim3 gemm_g_ff(DFF_   / TN, NROWS / TM);      // (32,16)
    const dim3 flash_g(S_LEN / 64, BATCH * NHEAD);      // (16,32)
    const float inv_scale = 0.125f;                     // 1/sqrt(HD)

    // xc = x
    copy_f32<<<(NROWS * DMODEL + 255) / 256, blk256, 0, stream>>>(x, xc, NROWS * DMODEL);

    for (int l = 0; l < NLAYER; ++l) {
        const float* Wq_l = Wq + (size_t)l * DMODEL * DMODEL;
        const float* Wk_l = Wk + (size_t)l * DMODEL * DMODEL;
        const float* Wv_l = Wv + (size_t)l * DMODEL * DMODEL;
        const float* Wo_l = Wo + (size_t)l * DMODEL * DMODEL;
        const float* W1_l = W1 + (size_t)l * DFF_ * DMODEL;
        const float* W2_l = W2 + (size_t)l * DMODEL * DFF_;
        const float* bq_l = bq + (size_t)l * DMODEL;
        const float* bk_l = bk + (size_t)l * DMODEL;
        const float* bv_l = bv + (size_t)l * DMODEL;
        const float* bo_l = bo + (size_t)l * DMODEL;
        const float* b1_l = b1 + (size_t)l * DFF_;
        const float* b2_l = b2 + (size_t)l * DMODEL;
        const float* rel_l = rel + (size_t)l * NUM_REL_ * HDIM;

        // Ebar for this layer
        rel_prefix<<<1, blk64, 0, stream>>>(rel_l, Pbuf);
        ebar_from_prefix<<<(S_LEN * HDIM) / 256, blk256, 0, stream>>>(Pbuf, ebar);

        // Q / Keff / V projections (permuted bf16 [B,H,S,HD] store)
        gemm_bf16_wmma<2><<<gemm_g_dm, blk256, 0, stream>>>(
            xc, Wq_l, bq_l, nullptr, qbuf, NROWS, DMODEL, DMODEL, 1.0f, (const float*)nullptr);
        gemm_bf16_wmma<2><<<gemm_g_dm, blk256, 0, stream>>>(
            xc, Wk_l, bk_l, nullptr, kbuf, NROWS, DMODEL, DMODEL, inv_scale, ebar);
        gemm_bf16_wmma<2><<<gemm_g_dm, blk256, 0, stream>>>(
            xc, Wv_l, bv_l, nullptr, vbuf, NROWS, DMODEL, DMODEL, 1.0f, (const float*)nullptr);

        // attention (TDM-staged K tiles)
        flash_attn_wmma<<<flash_g, blk128, 0, stream>>>(qbuf, kbuf, vbuf, attno);

        // output projection + residual LN
        gemm_bf16_wmma<0><<<gemm_g_dm, blk256, 0, stream>>>(
            attno, Wo_l, bo_l, tmp, nullptr, NROWS, DMODEL, DMODEL, 1.0f, (const float*)nullptr);
        add_layernorm<<<NROWS, blk256, 0, stream>>>(xc, tmp, g1 + (size_t)l * DMODEL, be1 + (size_t)l * DMODEL);

        // FFN: GELU(x W1^T + b1) W2^T + b2, residual LN
        gemm_bf16_wmma<1><<<gemm_g_ff, blk256, 0, stream>>>(
            xc, W1_l, b1_l, hbuf, nullptr, NROWS, DFF_, DMODEL, 1.0f, (const float*)nullptr);
        gemm_bf16_wmma<0><<<gemm_g_dm, blk256, 0, stream>>>(
            hbuf, W2_l, b2_l, tmp, nullptr, NROWS, DMODEL, DFF_, 1.0f, (const float*)nullptr);
        add_layernorm<<<NROWS, blk256, 0, stream>>>(xc, tmp, g2 + (size_t)l * DMODEL, be2 + (size_t)l * DMODEL);
    }

    copy_f32<<<(NROWS * DMODEL + 255) / 256, blk256, 0, stream>>>(xc, out, NROWS * DMODEL);
}
